// MultiHeadAttention_55250459296450
// MI455X (gfx1250) — matrix operand order repaired and checked by reading, not compiled, not run on hardware
//
#include <hip/hip_runtime.h>
#include <hip/hip_bf16.h>

// ---------------------------------------------------------------------------
// MHA with relative position embeddings, CDNA5 (gfx1250), wave32 + WMMA bf16.
// B=8, S=1024, E=1024, H=16, D=64, L=1024, SCALE=8
// CDNA5 data movement: double-buffered async global->LDS in the GEMMs
// (ASYNCcnt), TDM tensor_load_to_lds for K tiles in the flash kernel
// (TENSORcnt).
// ---------------------------------------------------------------------------

#define BB 8
#define SS 1024
#define EE 1024
#define HH 16
#define DD 64
#define LL 1024
#define PEVT_STRIDE 2064   // padded row stride for transposed pev table

typedef __attribute__((ext_vector_type(16))) __bf16 v16bf;
typedef __attribute__((ext_vector_type(8)))  __bf16 v8bf;
typedef __attribute__((ext_vector_type(8)))  float  v8f;
typedef __attribute__((ext_vector_type(4)))  int    v4i;
typedef __attribute__((ext_vector_type(8)))  int    v8i;
typedef __attribute__((ext_vector_type(4)))  unsigned int v4u;

#define WMMA_BF16(a, b, c) \
  __builtin_amdgcn_wmma_f32_16x16x32_bf16(false, (a), false, (b), (short)0, (c), false, false)

// ---- CDNA5 async global->LDS (ASYNCcnt) ----
#if __has_builtin(__builtin_amdgcn_global_load_async_to_lds_b128)
#define HAVE_ASYNC_LDS 1
typedef __attribute__((address_space(1))) v4i g_v4i;   // global (prints as __device__)
typedef __attribute__((address_space(3))) v4i l_v4i;   // LDS
#define ASYNC_COPY16(gp, lp) \
  __builtin_amdgcn_global_load_async_to_lds_b128((g_v4i*)(gp), (l_v4i*)(lp), 0, 0)
__device__ __forceinline__ void wait_async0() {
#if __has_builtin(__builtin_amdgcn_s_wait_asynccnt)
  __builtin_amdgcn_s_wait_asynccnt(0);
#else
  asm volatile("s_wait_asynccnt 0x0" ::: "memory");
#endif
}
#else
#define HAVE_ASYNC_LDS 0
#endif

// ---- CDNA5 TDM (TENSORcnt) ----
#if __has_builtin(__builtin_amdgcn_tensor_load_to_lds)
#define HAVE_TDM 1
__device__ __forceinline__ void wait_tensor0() {
#if __has_builtin(__builtin_amdgcn_s_wait_tensorcnt)
  __builtin_amdgcn_s_wait_tensorcnt(0);
#else
  asm volatile("s_wait_tensorcnt 0x0" ::: "memory");
#endif
}
#else
#define HAVE_TDM 0
#endif

__device__ __forceinline__ unsigned short f2bf(float f) {
  union { float f; unsigned u; } v; v.f = f;
  unsigned r = v.u + 0x7FFFu + ((v.u >> 16) & 1u);   // round-to-nearest-even
  return (unsigned short)(r >> 16);
}

__device__ __forceinline__ v16bf make_frag(v8bf lo, v8bf hi) {
  v16bf r;
#pragma unroll
  for (int i = 0; i < 8; i++) { r[i] = lo[i]; r[i + 8] = hi[i]; }
  return r;
}

// unaligned-safe 8x bf16 global load
__device__ __forceinline__ v8bf loadg8(const unsigned short* p) {
  v8bf r;
  __builtin_memcpy(&r, p, 16);
  return r;
}

// ---------------------------------------------------------------------------
// fp32 -> bf16 cast
// ---------------------------------------------------------------------------
__global__ void k_cvt_bf16(const float* __restrict__ in, unsigned short* __restrict__ out, int n) {
  int i = blockIdx.x * blockDim.x + threadIdx.x;
  if (i < n) out[i] = f2bf(in[i]);
}

// pev [2L+1][D] -> pevT [D][PEVT_STRIDE] (bf16), so B-fragments read contiguous u
__global__ void k_cvt_pevT(const float* __restrict__ pev, unsigned short* __restrict__ out) {
  int i = blockIdx.x * blockDim.x + threadIdx.x;
  if (i < (2 * LL + 1) * DD) {
    int u = i >> 6, d = i & 63;
    out[d * PEVT_STRIDE + u] = f2bf(pev[i]);
  }
}

// ---------------------------------------------------------------------------
// GEMM:  out[M,N] = A[M,K] @ W[N,K]^T   (torch Linear)
// 128x128 block tile, 256 threads (8 waves, 2x4), wave tile 64x32 (4x2 WMMA)
// Double-buffered async global->LDS pipeline when available.
// ---------------------------------------------------------------------------
__global__ __launch_bounds__(256)
void k_gemm_bf16(const unsigned short* __restrict__ A, const unsigned short* __restrict__ Bw,
                 unsigned short* __restrict__ outb, float* __restrict__ outf,
                 int M, int N, int K, int write_bf16) {
  __shared__ __attribute__((aligned(16))) unsigned short As[2][128][40];
  __shared__ __attribute__((aligned(16))) unsigned short Bs[2][128][40];

  const int tid  = threadIdx.x;
  const int lane = tid & 31;
  const int w    = tid >> 5;
  const int wm   = w >> 2;       // 0..1
  const int wn   = w & 3;        // 0..3
  const int lrow = lane & 15;
  const int lhi  = lane >> 4;
  const int m0 = blockIdx.y * 128, n0 = blockIdx.x * 128;

  v8f acc[4][2] = {};

  const int ldrow = tid >> 1;
  const int ldseg = (tid & 1) * 16;

  auto compute = [&](int buf) {
    v16bf af[4], bf[2];
#pragma unroll
    for (int mt = 0; mt < 4; mt++) {
      const unsigned short* p = &As[buf][wm * 64 + mt * 16 + lrow][lhi * 16];
      af[mt] = make_frag(*(const v8bf*)p, *(const v8bf*)(p + 8));
    }
#pragma unroll
    for (int nt = 0; nt < 2; nt++) {
      const unsigned short* p = &Bs[buf][wn * 32 + nt * 16 + lrow][lhi * 16];
      bf[nt] = make_frag(*(const v8bf*)p, *(const v8bf*)(p + 8));
    }
#pragma unroll
    for (int mt = 0; mt < 4; mt++)
#pragma unroll
      for (int nt = 0; nt < 2; nt++)
        acc[mt][nt] = WMMA_BF16(af[mt], bf[nt], acc[mt][nt]);
  };

#if HAVE_ASYNC_LDS
  auto issue = [&](int buf, int k0) {
    const unsigned short* ga = A  + (size_t)(m0 + ldrow) * K + k0 + ldseg;
    const unsigned short* gb = Bw + (size_t)(n0 + ldrow) * K + k0 + ldseg;
    ASYNC_COPY16(ga,     &As[buf][ldrow][ldseg]);
    ASYNC_COPY16(ga + 8, &As[buf][ldrow][ldseg + 8]);
    ASYNC_COPY16(gb,     &Bs[buf][ldrow][ldseg]);
    ASYNC_COPY16(gb + 8, &Bs[buf][ldrow][ldseg + 8]);
  };
  const int nk = K >> 5;
  issue(0, 0);
  for (int kt = 0; kt < nk; kt++) {
    const int cur = kt & 1;
    wait_async0();        // my tile landed in LDS
    __syncthreads();      // everyone's tile landed; prev compute done
    if (kt + 1 < nk) issue(cur ^ 1, (kt + 1) << 5);
    compute(cur);
  }
#else
  for (int k0 = 0; k0 < K; k0 += 32) {
    const unsigned short* ga = A  + (size_t)(m0 + ldrow) * K + k0 + ldseg;
    const unsigned short* gb = Bw + (size_t)(n0 + ldrow) * K + k0 + ldseg;
    __builtin_prefetch(ga + 32, 0, 0);
    __builtin_prefetch(gb + 32, 0, 0);
    *(uint4*)&As[0][ldrow][ldseg]     = *(const uint4*)ga;
    *(uint4*)&As[0][ldrow][ldseg + 8] = *(const uint4*)(ga + 8);
    *(uint4*)&Bs[0][ldrow][ldseg]     = *(const uint4*)gb;
    *(uint4*)&Bs[0][ldrow][ldseg + 8] = *(const uint4*)(gb + 8);
    __syncthreads();
    compute(0);
    __syncthreads();
  }
#endif

#pragma unroll
  for (int mt = 0; mt < 4; mt++)
#pragma unroll
    for (int nt = 0; nt < 2; nt++) {
      int nn = n0 + wn * 32 + nt * 16 + lrow;
      int mb = m0 + wm * 64 + mt * 16 + lhi * 8;
#pragma unroll
      for (int i = 0; i < 8; i++) {
        size_t idx = (size_t)(mb + i) * N + nn;
        if (write_bf16) outb[idx] = f2bf(acc[mt][nt][i]);
        else            outf[idx] = acc[mt][nt][i];
      }
    }
}

// ---------------------------------------------------------------------------
// Fused flash attention + RPE.
// grid: (S/64, B*H); block: 128 threads = 4 waves; wave owns 16 query rows.
// K tile is fetched by the Tensor Data Mover (64x64 bf16 2D tile, row stride
// 1024 elem, LDS pad 16B per 128B row -> 72-elem LDS stride) when available.
// ---------------------------------------------------------------------------
__global__ __launch_bounds__(128)
void k_flash_rpe(const unsigned short* __restrict__ Qb, const unsigned short* __restrict__ Kb,
                 const unsigned short* __restrict__ Vb, const unsigned short* __restrict__ pekb,
                 const unsigned short* __restrict__ pevT, float* __restrict__ Of) {
  __shared__ __attribute__((aligned(16))) unsigned short Kt[64][72];   // [key][d]
  __shared__ __attribute__((aligned(16))) unsigned short Vt[64][72];   // [d][key] (transposed)
  __shared__ __attribute__((aligned(16))) unsigned short Pt[4][16][72];   // probs [row][key]
  __shared__ __attribute__((aligned(16))) unsigned short Ash[4][16][104]; // shifted probs [row][u]
  __shared__ float Qp[4][16][96];

  const int tid  = threadIdx.x;
  const int lane = tid & 31;
  const int w    = tid >> 5;
  const int lrow = lane & 15;
  const int lhi  = lane >> 4;

  const int bh = blockIdx.y;
  const int b  = bh >> 4;          // /H
  const int h  = bh & 15;
  const int it = blockIdx.x;
  const int s0 = it * 64;
  const int sw = s0 + w * 16;

  // scrambled q index for the RPE score term: n = h*B + b ; (bq,hq) = (n/H, n%H)
  const int n  = h * BB + b;
  const int bq = n >> 4;
  const int hq = n & 15;

  // Q fragments (regular + scrambled), rows sw..sw+15, D=64 -> 2 k-steps
  v16bf qa[2], qs[2];
  {
    const int s = sw + lrow;
    const unsigned short* qp  = Qb + ((size_t)(b  * SS + s) * EE + h  * DD);
    const unsigned short* qsp = Qb + ((size_t)(bq * SS + s) * EE + hq * DD);
#pragma unroll
    for (int kk = 0; kk < 2; kk++) {
      int c0 = kk * 32 + lhi * 16;
      qa[kk] = make_frag(*(const v8bf*)(qp + c0),  *(const v8bf*)(qp + c0 + 8));
      qs[kk] = make_frag(*(const v8bf*)(qsp + c0), *(const v8bf*)(qsp + c0 + 8));
    }
  }

  v8f oacc[4] = {};
  float rm[8], rl[8];
#pragma unroll
  for (int i = 0; i < 8; i++) { rm[i] = -1e30f; rl[i] = 0.f; }

  for (int jt = 0; jt <= it; jt++) {
    const int t0 = jt * 64;
    __syncthreads();

#if HAVE_TDM
    if (w == 0) {  // one TDM descriptor per workgroup fetches the K tile
      unsigned long long ga = (unsigned long long)(Kb + (size_t)(b * SS + t0) * EE + h * DD);
      unsigned lds = (unsigned)(size_t)(&Kt[0][0]);
      v4u g0;
      g0[0] = 1u;                                          // count=1, user descriptor
      g0[1] = lds;                                         // lds_addr
      g0[2] = (unsigned)(ga & 0xFFFFFFFFu);                // global_addr[31:0]
      g0[3] = (unsigned)((ga >> 32) & 0x01FFFFFFu) | (2u << 30);  // addr[56:32] | type=2
      v8i g1 = (v8i)0;
      g1[0] = (1 << 16)      // data_size = 2 bytes
            | (1 << 20)      // pad_enable
            | (4 << 22)      // pad_interval: 32 DW (128B) between pads
            | (3 << 25);     // pad_amount: 4 DW (16B)
      g1[1] = (int)(64u << 16);   // tensor_dim0 = 64 (low 16 bits at [63:48])
      g1[2] = (int)(64u << 16);   // tensor_dim0 hi=0 | tensor_dim1 = 64
      g1[3] = (int)(64u << 16);   // tensor_dim1 hi=0 | tile_dim0 = 64
      g1[4] = 64;                 // tile_dim1 = 64 | tile_dim2 = 0
      g1[5] = 1024;               // tensor_dim0_stride = 1024 elements
      g1[6] = 0;
      g1[7] = 0;
      v4i gz = (v4i)0;
#if __clang_major__ >= 23
      v8i gz8 = (v8i)0;
      __builtin_amdgcn_tensor_load_to_lds(g0, g1, gz, gz, gz8, 0);
#else
      __builtin_amdgcn_tensor_load_to_lds(g0, g1, gz, gz, 0);
#endif
      wait_tensor0();
    }
    {  // V^T tile: regular loads + transposed LDS scatter (TDM cannot transpose)
      int row = tid & 63, half = tid >> 6;
      const unsigned short* vg = Vb + ((size_t)(b * SS + t0 + row) * EE + h * DD + half * 32);
#pragma unroll
      for (int q = 0; q < 32; q++) Vt[half * 32 + q][row] = vg[q];
    }
#else
    {  // cooperative K / V^T tile load (128 threads)
      int row = tid & 63, half = tid >> 6;
      const unsigned short* kg = Kb + ((size_t)(b * SS + t0 + row) * EE + h * DD + half * 32);
      *(uint4*)&Kt[row][half * 32]      = *(const uint4*)kg;
      *(uint4*)&Kt[row][half * 32 + 8]  = *(const uint4*)(kg + 8);
      *(uint4*)&Kt[row][half * 32 + 16] = *(const uint4*)(kg + 16);
      *(uint4*)&Kt[row][half * 32 + 24] = *(const uint4*)(kg + 24);
      const unsigned short* vg = Vb + ((size_t)(b * SS + t0 + row) * EE + h * DD + half * 32);
#pragma unroll
      for (int q = 0; q < 32; q++) Vt[half * 32 + q][row] = vg[q];
    }
#endif
    // zero this wave's shifted-prob matrix
    for (int e = lane; e < 16 * 96; e += 32) Ash[w][e / 96][e % 96] = 0;
    __syncthreads();

    // ---- scores: Q @ K^T ----
    v8f sacc[4] = {};
#pragma unroll
    for (int nt = 0; nt < 4; nt++)
#pragma unroll
      for (int kk = 0; kk < 2; kk++) {
        const unsigned short* p = &Kt[nt * 16 + lrow][kk * 32 + lhi * 16];
        v16bf bk = make_frag(*(const v8bf*)p, *(const v8bf*)(p + 8));
        sacc[nt] = WMMA_BF16(qa[kk], bk, sacc[nt]);
      }

    // ---- RPE score: Qp[s,u] = q' . pek_table[u0+u], u-window of 96 ----
    const int u0 = t0 - sw - 15 + LL;   // always in [1, 1009]
    v8f pacc[6] = {};
#pragma unroll
    for (int ut = 0; ut < 6; ut++)
#pragma unroll
      for (int kk = 0; kk < 2; kk++) {
        const unsigned short* p = pekb + (size_t)(u0 + ut * 16 + lrow) * DD + kk * 32 + lhi * 16;
        v16bf be = make_frag(*(const v8bf*)p, *(const v8bf*)(p + 8));
        pacc[ut] = WMMA_BF16(qs[kk], be, pacc[ut]);
      }
#pragma unroll
    for (int ut = 0; ut < 6; ut++)
#pragma unroll
      for (int i = 0; i < 8; i++)
        Qp[w][lhi * 8 + i][ut * 16 + lrow] = pacc[ut][i];
    __syncthreads();

    // ---- combine + causal mask + online softmax ----
    float pv[4][8];
#pragma unroll
    for (int i = 0; i < 8; i++) {
      const int r = lhi * 8 + i;
      const int s = sw + r;
      float vals[4];
      float mloc = -1e30f;
#pragma unroll
      for (int nt = 0; nt < 4; nt++) {
        int t = t0 + nt * 16 + lrow;
        float vv = (sacc[nt][i] + Qp[w][r][nt * 16 + lrow - r + 15]) * 0.125f;
        if (t > s) vv = -1e30f;
        vals[nt] = vv;
        mloc = fmaxf(mloc, vv);
      }
#pragma unroll
      for (int off = 1; off < 16; off <<= 1)
        mloc = fmaxf(mloc, __shfl_xor(mloc, off, 32));
      float nm    = fmaxf(rm[i], mloc);
      float scale = __expf(rm[i] - nm);
      float ssum  = 0.f;
#pragma unroll
      for (int nt = 0; nt < 4; nt++) {
        float p = __expf(vals[nt] - nm);
        pv[nt][i] = p;
        ssum += p;
      }
#pragma unroll
      for (int off = 1; off < 16; off <<= 1)
        ssum += __shfl_xor(ssum, off, 32);
      rl[i] = rl[i] * scale + ssum;
      rm[i] = nm;
#pragma unroll
      for (int nt = 0; nt < 4; nt++) oacc[nt][i] *= scale;
    }

    // stage probabilities (and shifted copy for the value-RPE GEMM)
#pragma unroll
    for (int i = 0; i < 8; i++) {
      int r = lhi * 8 + i;
#pragma unroll
      for (int nt = 0; nt < 4; nt++) {
        unsigned short pb = f2bf(pv[nt][i]);
        int tl = nt * 16 + lrow;
        Pt[w][r][tl]            = pb;
        Ash[w][r][tl - r + 15]  = pb;   // column = (t - s + L) - u0, in [0,78]
      }
    }
    __syncthreads();

    // ---- O += P @ V ----
#pragma unroll
    for (int kk = 0; kk < 2; kk++) {
      const unsigned short* pA = &Pt[w][lrow][kk * 32 + lhi * 16];
      v16bf ap = make_frag(*(const v8bf*)pA, *(const v8bf*)(pA + 8));
#pragma unroll
      for (int nt = 0; nt < 4; nt++) {
        const unsigned short* pB = &Vt[nt * 16 + lrow][kk * 32 + lhi * 16];
        v16bf bv = make_frag(*(const v8bf*)pB, *(const v8bf*)(pB + 8));
        oacc[nt] = WMMA_BF16(ap, bv, oacc[nt]);
      }
    }
    // ---- O += Ash @ pevT  (96-deep in u) ----
#pragma unroll
    for (int kk = 0; kk < 3; kk++) {
      const unsigned short* pA = &Ash[w][lrow][kk * 32 + lhi * 16];
      v16bf ap = make_frag(*(const v8bf*)pA, *(const v8bf*)(pA + 8));
#pragma unroll
      for (int nt = 0; nt < 4; nt++) {
        const unsigned short* pB = pevT + (size_t)(nt * 16 + lrow) * PEVT_STRIDE
                                        + u0 + kk * 32 + lhi * 16;
        v16bf bv = make_frag(loadg8(pB), loadg8(pB + 8));  // u0 unaligned
        oacc[nt] = WMMA_BF16(ap, bv, oacc[nt]);
      }
    }
  }

  // epilogue: normalize and write O as [B,S,E] fp32
#pragma unroll
  for (int nt = 0; nt < 4; nt++)
#pragma unroll
    for (int i = 0; i < 8; i++) {
      int r = lhi * 8 + i;
      int s = sw + r;
      int d = nt * 16 + lrow;
      Of[(size_t)(b * SS + s) * EE + h * DD + d] = oacc[nt][i] / rl[i];
    }
}

// ---------------------------------------------------------------------------
extern "C" void kernel_launch(void* const* d_in, const int* in_sizes, int n_in,
                              void* d_out, int out_size, void* d_ws, size_t ws_size,
                              hipStream_t stream) {
  const float* x   = (const float*)d_in[0];
  // d_in[1] = mask (causal, implicit in kernel)
  const float* Wq  = (const float*)d_in[2];
  const float* Wk  = (const float*)d_in[3];
  const float* Wv  = (const float*)d_in[4];
  const float* Wo  = (const float*)d_in[5];
  const float* pek = (const float*)d_in[6];
  const float* pev = (const float*)d_in[7];

  char* ws = (char*)d_ws;
  size_t off = 0;
  auto alloc = [&](size_t bytes) -> void* {
    void* p = ws + off;
    off += (bytes + 255) & ~(size_t)255;
    return p;
  };

  const size_t BS = (size_t)BB * SS;          // 8192
  unsigned short* xb    = (unsigned short*)alloc(BS * EE * 2);
  unsigned short* Wqb   = (unsigned short*)alloc((size_t)EE * EE * 2);
  unsigned short* Wkb   = (unsigned short*)alloc((size_t)EE * EE * 2);
  unsigned short* Wvb   = (unsigned short*)alloc((size_t)EE * EE * 2);
  unsigned short* Wob   = (unsigned short*)alloc((size_t)EE * EE * 2);
  unsigned short* pekb  = (unsigned short*)alloc((size_t)(2 * LL + 1) * DD * 2);
  unsigned short* pevTb = (unsigned short*)alloc((size_t)DD * PEVT_STRIDE * 2);
  unsigned short* Qb    = (unsigned short*)alloc(BS * EE * 2);
  unsigned short* Kb    = (unsigned short*)alloc(BS * EE * 2);
  unsigned short* Vb    = (unsigned short*)alloc(BS * EE * 2);
  float*          Ofp   = (float*)         alloc(BS * EE * 4);
  unsigned short* Ofb   = (unsigned short*)alloc(BS * EE * 2);

  const int nx = (int)(BS * EE);
  const int nw = EE * EE;
  const int np = (2 * LL + 1) * DD;

  k_cvt_bf16<<<(nx + 255) / 256, 256, 0, stream>>>(x, xb, nx);
  k_cvt_bf16<<<(nw + 255) / 256, 256, 0, stream>>>(Wq, Wqb, nw);
  k_cvt_bf16<<<(nw + 255) / 256, 256, 0, stream>>>(Wk, Wkb, nw);
  k_cvt_bf16<<<(nw + 255) / 256, 256, 0, stream>>>(Wv, Wvb, nw);
  k_cvt_bf16<<<(nw + 255) / 256, 256, 0, stream>>>(Wo, Wob, nw);
  k_cvt_bf16<<<(np + 255) / 256, 256, 0, stream>>>(pek, pekb, np);
  k_cvt_pevT<<<(np + 255) / 256, 256, 0, stream>>>(pev, pevTb);

  dim3 gg(EE / 128, (int)(BS / 128));
  k_gemm_bf16<<<gg, 256, 0, stream>>>(xb, Wqb, Qb, nullptr, (int)BS, EE, EE, 1);
  k_gemm_bf16<<<gg, 256, 0, stream>>>(xb, Wkb, Kb, nullptr, (int)BS, EE, EE, 1);
  k_gemm_bf16<<<gg, 256, 0, stream>>>(xb, Wvb, Vb, nullptr, (int)BS, EE, EE, 1);

  k_flash_rpe<<<dim3(SS / 64, BB * HH), 128, 0, stream>>>(Qb, Kb, Vb, pekb, pevTb, Ofp);

  k_cvt_bf16<<<(nx + 255) / 256, 256, 0, stream>>>(Ofp, Ofb, nx);
  k_gemm_bf16<<<gg, 256, 0, stream>>>(Ofb, Wob, nullptr, (float*)d_out, (int)BS, EE, EE, 0);
}
